// MultiAgentActorCritic_43834436223483
// MI455X (gfx1250) — compile-verified, matrix operand order checked by reading, not run on hardware
//
#include <hip/hip_runtime.h>
#include <hip/hip_bf16.h>
#include <math.h>

// ---------------- types for WMMA ----------------
typedef __attribute__((ext_vector_type(16))) __bf16 v16bf;
typedef __attribute__((ext_vector_type(8)))  __bf16 v8bf;
typedef __attribute__((ext_vector_type(8)))  float  v8f;

#define B_   1024
#define N_   64
#define OBS_ 128
#define ACT_ 16
#define HID_ 256
#define GD_  64
#define H_   4
#define AI_  (OBS_ + GD_)    // 192
#define CI_  (OBS_ + ACT_)   // 144

// Assemble a 16-element bf16 fragment from an LDS row laid out with the
// ISA 7.12.2 16-bit operand K-mapping: lanes 0-15 take K = {0..7,16..23},
// lanes 16-31 take K = {8..15,24..31}. Two aligned 16B vector loads.
__device__ __forceinline__ v16bf frag_ld(const __bf16* p, int hi) {
  v8bf a = *(const v8bf*)(p + hi * 8);
  v8bf b = *(const v8bf*)(p + 16 + hi * 8);
  return __builtin_shufflevector(a, b, 0, 1, 2, 3, 4, 5, 6, 7,
                                 8, 9, 10, 11, 12, 13, 14, 15);
}

// branchless ELU using the hardware exp (v_exp_f32)
__device__ __forceinline__ float elu_fast(float v) {
  float ev = __expf(fminf(v, 0.0f)) - 1.0f;     // exp clamped: no range fixup
  return v > 0.0f ? v : ev;                     // v_cndmask, no exec juggling
}

// ---------------------------------------------------------------------------
// LDS-staged batched GEMM: C[b] = act(A[b] (MxK) * B[b] (KxN) + bias[b]).
// Requirements (all call sites comply): M % 64 == 0, K % 16 == 0, and all
// intra-batch extents fit 32-bit (row*ld < 2^31).
// Block: 256 threads -> 64x64 C tile; 8 waves x 2 wmma tiles; K-step 32 via
// v_wmma_f32_16x16x32_bf16 (f32 accumulate). B staged transposed so both
// operands are fragment-loaded from LDS with aligned 16B vector reads.
// EDGE=false fast path (K%32==0 && Nc%64==0): zero guards in the K-loop,
// next-panel global_prefetch. EDGE=true handles K tail + partial N tiles.
// act: 0 = none, 1 = relu
// ---------------------------------------------------------------------------
template <bool EDGE>
__global__ __launch_bounds__(256) void gemm_wmma_bf16(
    const float* __restrict__ A, const float* __restrict__ Bm,
    const float* __restrict__ bias, float* __restrict__ C,
    int M, int Nc, int K,
    int lda, int ldb, int ldc,
    long long sA, long long sB, long long sC, long long sBias,
    int tilesM, int tilesN, int act)
{
  __shared__ alignas(16) __bf16 lA[64][40];   // [m][k]   row = 80B (16B mult)
  __shared__ alignas(16) __bf16 lB[64][40];   // [n][k]   transposed B panel

  int tpb  = tilesM * tilesN;
  int bidx = blockIdx.x / tpb;
  int rem  = blockIdx.x % tpb;
  int bm = rem / tilesN;
  int bn = rem % tilesN;

  const float* Ab = A  + (long long)bidx * sA + (long long)bm * 64 * lda;
  const float* Bb = Bm + (long long)bidx * sB + bn * 64;
  int ncols = Nc - bn * 64; if (ncols > 64) ncols = 64;

  int t = threadIdx.x;
  int wave = t >> 5, lane = t & 31;
  int r = lane & 15, hi = lane >> 4;
  int tm  = wave & 3;        // this wave's M tile (0..3)
  int tnA = wave >> 2;       // this wave's two N tiles
  int tnB = tnA + 2;

  v8f acc0 = {}, acc1 = {};

  for (int k0 = 0; k0 < K; k0 += 32) {
    // ---- stage A panel (64 x 32) as bf16, float4 global loads ----
#pragma unroll
    for (int it = 0; it < 2; ++it) {
      int i   = it * 256 + t;        // 512 float4 slots
      int row = i >> 3;
      int kq  = (i & 7) * 4;
      int k   = k0 + kq;
      float4 v;
      if (EDGE) {
        v = make_float4(0.f, 0.f, 0.f, 0.f);
        if (k < K)                   // K % 16 == 0 -> whole float4 in/out
          v = *(const float4*)(Ab + row * lda + k);
      } else {
        const float* ap = Ab + row * lda + k;
        v = *(const float4*)ap;
        if (k0 + 32 < K) __builtin_prefetch(ap + 32, 0, 1);  // next K panel
      }
      __bf16* d = &lA[row][kq];
      d[0] = (__bf16)v.x; d[1] = (__bf16)v.y;
      d[2] = (__bf16)v.z; d[3] = (__bf16)v.w;
    }
    // ---- stage B panel (32 k x 64 n) transposed -> lB[n][k] ----
#pragma unroll
    for (int it = 0; it < 2; ++it) {
      int i  = it * 256 + t;
      int kr = i >> 4;               // 0..31
      int nq = (i & 15) * 4;         // 0..60
      int kk = k0 + kr;
      float4 v;
      if (EDGE) {
        v = make_float4(0.f, 0.f, 0.f, 0.f);
        if (kk < K) {
          const float* p = Bb + kk * ldb;
          if (nq + 3 < ncols) {
            v = *(const float4*)(p + nq);
          } else if (nq < ncols) {               // N edge (Nc=1/16 cases)
            v.x = p[nq];
            if (nq + 1 < ncols) v.y = p[nq + 1];
            if (nq + 2 < ncols) v.z = p[nq + 2];
          }
        }
      } else {
        const float* bp = Bb + kk * ldb + nq;
        v = *(const float4*)bp;
        if (k0 + 32 < K) __builtin_prefetch(bp + 32 * ldb, 0, 1);
      }
      lB[nq + 0][kr] = (__bf16)v.x;
      lB[nq + 1][kr] = (__bf16)v.y;
      lB[nq + 2][kr] = (__bf16)v.z;
      lB[nq + 3][kr] = (__bf16)v.w;
    }
    __syncthreads();

    v16bf af = frag_ld(&lA[tm  * 16 + r][0], hi);
    v16bf b0 = frag_ld(&lB[tnA * 16 + r][0], hi);
    v16bf b1 = frag_ld(&lB[tnB * 16 + r][0], hi);
    acc0 = __builtin_amdgcn_wmma_f32_16x16x32_bf16(
        false, af, false, b0, (short)0, acc0, false, false);
    acc1 = __builtin_amdgcn_wmma_f32_16x16x32_bf16(
        false, af, false, b1, (short)0, acc1, false, false);
    __syncthreads();
  }

  // ---- epilogue: bias + activation + stores (32-bit intra-batch math) ----
  float* Cb = C + (long long)bidx * sC;
  const float* biasb = bias ? (bias + (long long)bidx * sBias) : nullptr;
  int gnA = bn * 64 + tnA * 16 + r;
  int gnB = bn * 64 + tnB * 16 + r;
  float bbA = 0.f, bbB = 0.f;
  if (biasb) {
    if (!EDGE || gnA < Nc) bbA = biasb[gnA];
    if (!EDGE || gnB < Nc) bbB = biasb[gnB];
  }
  int rbase = (bm * 64 + tm * 16 + 8 * hi) * ldc;   // < 2^31 by construction
#pragma unroll
  for (int vi = 0; vi < 8; ++vi) {
    int o = rbase + vi * ldc;
    float v0 = acc0[vi] + bbA;
    float v1 = acc1[vi] + bbB;
    if (act == 1) { v0 = fmaxf(v0, 0.f); v1 = fmaxf(v1, 0.f); }
    if (!EDGE || gnA < Nc) Cb[o + gnA] = v0;
    if (!EDGE || gnB < Nc) Cb[o + gnB] = v1;
  }
}

// ---------------------------------------------------------------------------
// GAT attention head: scores = leaky_relu(si+sj) + adj mask + row softmax in
// LDS (f32), bf16 weight & transposed-xt panels in LDS, then
// out = softmax(scores) @ xt (64x64 @ 64x64) via WMMA. blockIdx.x = b*H + h.
// act: 0 = none, 2 = ELU
// ---------------------------------------------------------------------------
__global__ __launch_bounds__(256) void gat_attn(
    const float* __restrict__ s1, const float* __restrict__ s2,
    const float* __restrict__ adj, const float* __restrict__ xt,
    float* __restrict__ out, long long hStride, int ldo, int act)
{
  int bh = blockIdx.x;
  int b = bh >> 2;
  int h = bh & 3;
  __shared__ float sc[64][72];                   // f32 scores / softmax
  __shared__ alignas(16) __bf16 lW[64][72];      // bf16 attn weights [i][j]
  __shared__ alignas(16) __bf16 lX[64][72];      // bf16 xt transposed [d][k]
  int t = threadIdx.x;

  const float* s1p = s1 + ((long long)h * B_ + b) * N_;
  const float* s2p = s2 + ((long long)h * B_ + b) * N_;
  const float* adp = adj + (long long)b * N_ * N_;
  const float* xtp = xt + (((long long)h * B_ + b) * N_) * GD_;
  float* outp = out + (long long)h * hStride + (long long)b * N_ * ldo;

  for (int idx = t; idx < N_ * N_; idx += 256) {
    int i = idx >> 6, j = idx & 63;
    float v = s1p[i] + s2p[j];
    v = v > 0.0f ? v : 0.2f * v;                 // leaky relu alpha=0.2
    if (adp[idx] == 0.0f) v = -INFINITY;
    sc[i][j] = v;
    lX[j][i] = (__bf16)xtp[idx];                 // idx = k*64 + d -> lX[d][k]
  }
  __syncthreads();

  if (t < 64) {                                  // row softmax -> bf16 weights
    float m = -INFINITY;
    for (int j = 0; j < 64; ++j) m = fmaxf(m, sc[t][j]);
    float ssum = 0.0f;
    for (int j = 0; j < 64; ++j) { float e = __expf(sc[t][j] - m); sc[t][j] = e; ssum += e; }
    float inv = 1.0f / ssum;
    for (int j = 0; j < 64; ++j) lW[t][j] = (__bf16)(sc[t][j] * inv);
  }
  __syncthreads();

  int wave = t >> 5, lane = t & 31;
  int r = lane & 15, hi = lane >> 4;

  for (int tile = wave; tile < 16; tile += 8) {
    int ti = tile >> 2, td = tile & 3;
    v8f acc = {};
#pragma unroll
    for (int ks = 0; ks < 64; ks += 32) {
      v16bf af  = frag_ld(&lW[ti * 16 + r][ks], hi);
      v16bf bfr = frag_ld(&lX[td * 16 + r][ks], hi);
      acc = __builtin_amdgcn_wmma_f32_16x16x32_bf16(
          false, af, false, bfr, (short)0, acc, false, false);
    }
    int obase = (ti * 16 + 8 * hi) * ldo + td * 16 + r;   // 32-bit index math
#pragma unroll
    for (int vi = 0; vi < 8; ++vi) {
      float v = acc[vi];
      if (act == 2) v = elu_fast(v);
      outp[obase + vi * ldo] = v;
    }
  }
}

// si = xt . a[:GD], sj = xt . a[GD:]  per (h,b,n)
__global__ void gat_scores(const float* __restrict__ xt,
                           const float* __restrict__ a,
                           float* __restrict__ s1, float* __restrict__ s2)
{
  long long idx = (long long)blockIdx.x * 256 + threadIdx.x;
  long long total = (long long)H_ * B_ * N_;
  if (idx >= total) return;
  int h = (int)(idx >> 16);                      // B_*N_ == 65536
  const float* xp = xt + idx * GD_;
  const float* a1 = a + h * 2 * GD_;
  const float* a2 = a1 + GD_;
  float v1 = 0.0f, v2 = 0.0f;
#pragma unroll 4
  for (int d = 0; d < GD_; ++d) { float x = xp[d]; v1 += x * a1[d]; v2 += x * a2[d]; }
  s1[idx] = v1; s2[idx] = v2;
}

// out[bn,d] (=|+=) mean_h at[h,bn,d]
__global__ void head_mean(const float* __restrict__ at, float* __restrict__ out,
                          int addFlag)
{
  long long idx = (long long)blockIdx.x * 256 + threadIdx.x;
  long long total = (long long)B_ * N_ * GD_;
  if (idx >= total) return;
  long long S = total;
  float v = 0.25f * (at[idx] + at[S + idx] + at[2 * S + idx] + at[3 * S + idx]);
  if (addFlag) out[idx] += v; else out[idx] = v;
}

// one block per row: out[row] = [in1[row] | in2[row]]   (no int division)
__global__ void concat2(const float* __restrict__ in1, const float* __restrict__ in2,
                        float* __restrict__ out, int c1, int c2)
{
  long long row = blockIdx.x;
  int ct = c1 + c2;
  const float* p1 = in1 + row * c1;
  const float* p2 = in2 + row * c2;
  float* q = out + row * ct;
  for (int c = threadIdx.x; c < ct; c += blockDim.x)
    q[c] = (c < c1) ? p1[c] : p2[c - c1];
}

__global__ void add_vec(const float* __restrict__ a, const float* __restrict__ b,
                        float* __restrict__ out, long long n)
{
  long long idx = (long long)blockIdx.x * 256 + threadIdx.x;
  if (idx >= n) return;
  out[idx] = a[idx] + b[idx];
}

// row-wise softmax for small matrices (one thread per row)
__global__ void row_softmax(const float* __restrict__ in, float* __restrict__ out,
                            int rows, int cols)
{
  int rr = blockIdx.x * blockDim.x + threadIdx.x;
  if (rr >= rows) return;
  const float* p = in + (long long)rr * cols;
  float* q = out + (long long)rr * cols;
  float m = -INFINITY;
  for (int c = 0; c < cols; ++c) m = fmaxf(m, p[c]);
  float s = 0.0f;
  for (int c = 0; c < cols; ++c) { float e = __expf(p[c] - m); q[c] = e; s += e; }
  float inv = 1.0f / s;
  for (int c = 0; c < cols; ++c) q[c] *= inv;
}

// ---------------------------------------------------------------------------
extern "C" void kernel_launch(void* const* d_in, const int* in_sizes, int n_in,
                              void* d_out, int out_size, void* d_ws, size_t ws_size,
                              hipStream_t stream) {
  (void)in_sizes; (void)n_in; (void)out_size; (void)ws_size;
  const float* obs   = (const float*)d_in[0];
  const float* acts  = (const float*)d_in[1];
  const float* adj   = (const float*)d_in[2];
  const float* caus  = (const float*)d_in[3];
  const float* gatW  = (const float*)d_in[4];
  const float* gatA  = (const float*)d_in[5];
  const float* cgW0  = (const float*)d_in[6];
  const float* cgA0  = (const float*)d_in[7];
  const float* cgW1  = (const float*)d_in[8];
  const float* cgA1  = (const float*)d_in[9];
  const float* aw1 = (const float*)d_in[10]; const float* ab1 = (const float*)d_in[11];
  const float* aw2 = (const float*)d_in[12]; const float* ab2 = (const float*)d_in[13];
  const float* aw3 = (const float*)d_in[14]; const float* ab3 = (const float*)d_in[15];
  const float* kw1 = (const float*)d_in[16]; const float* kb1 = (const float*)d_in[17];
  const float* kw2 = (const float*)d_in[18]; const float* kb2 = (const float*)d_in[19];
  const float* kw3 = (const float*)d_in[20]; const float* kb3 = (const float*)d_in[21];
  const float* cw1 = (const float*)d_in[22]; const float* cb1 = (const float*)d_in[23];
  const float* cw2 = (const float*)d_in[24]; const float* cb2 = (const float*)d_in[25];
  const float* cw3 = (const float*)d_in[26]; const float* cb3 = (const float*)d_in[27];
  const float* scmC = (const float*)d_in[28];
  const float* mw1 = (const float*)d_in[29]; const float* mb1 = (const float*)d_in[30];
  const float* mw2 = (const float*)d_in[31]; const float* mb2 = (const float*)d_in[32];
  const float* mw3 = (const float*)d_in[33]; const float* mb3 = (const float*)d_in[34];
  const float* nw1 = (const float*)d_in[35]; const float* nb1 = (const float*)d_in[36];
  const float* nw2 = (const float*)d_in[37]; const float* nb2 = (const float*)d_in[38];

  float* outF = (float*)d_out;
  float* out_actor  = outF;                                   // (B,N,ACT)
  float* out_critic = outF + (long long)B_ * N_ * ACT_;       // (B,N,1)
  float* out_cent   = out_critic + (long long)B_ * N_;        // (B,N)
  float* out_scm    = out_cent + (long long)B_ * N_;          // (B,N,OBS)

  // ---- workspace bump allocator ----
  size_t off = 0;
  auto alloc = [&](size_t nflt) {
    float* p = (float*)((char*)d_ws + off);
    off += ((nflt * sizeof(float)) + 255) & ~(size_t)255;
    return p;
  };
  const long long HBNG = (long long)H_ * B_ * N_ * GD_;
  float* smxC = alloc((size_t)OBS_ * OBS_);
  float* smxN = alloc((size_t)N_ * N_);
  float* s1b  = alloc((size_t)H_ * B_ * N_);
  float* s2b  = alloc((size_t)H_ * B_ * N_);
  float* xt   = alloc((size_t)HBNG);                          // head projections
  float* at   = alloc((size_t)HBNG);                          // per-head attn out
  float* hcg  = alloc((size_t)B_ * N_ * (H_ * GD_));          // concat GAT layer
  float* xm   = alloc((size_t)B_ * N_ * OBS_);
  float* comm = alloc((size_t)B_ * N_ * GD_);
  float* h1   = alloc((size_t)B_ * N_ * HID_);
  float* h2   = alloc((size_t)B_ * N_ * HID_);
  float* cmb  = alloc((size_t)B_ * N_ * CI_);                 // obs||actions
  float* c1   = alloc((size_t)B_ * HID_);
  float* c2   = alloc((size_t)B_ * HID_);
  // overlays (disjoint lifetimes, single stream = sequential):
  float* ai    = hcg;   // (B,N,AI)   after hcg consumed
  float* ce    = xm;    // (B,N,OBS)  after xm consumed
  float* se    = at;    // (B,N,OBS)  after at consumed
  float* noise = xt;    // (B,N,OBS)  after xt consumed

  auto gemm = [&](const float* A, const float* Bm, const float* bias, float* C,
                  int M, int Nc, int K,
                  int lda, int ldb, int ldc,
                  long long sA, long long sB, long long sC, long long sBias,
                  int batch, int act) {
    int tm = (M + 63) / 64, tn = (Nc + 63) / 64;
    int blocks = batch * tm * tn;
    if ((K % 32) == 0 && (Nc % 64) == 0)
      gemm_wmma_bf16<false><<<blocks, 256, 0, stream>>>(
          A, Bm, bias, C, M, Nc, K, lda, ldb, ldc, sA, sB, sC, sBias, tm, tn, act);
    else
      gemm_wmma_bf16<true><<<blocks, 256, 0, stream>>>(
          A, Bm, bias, C, M, Nc, K, lda, ldb, ldc, sA, sB, sC, sBias, tm, tn, act);
  };
  auto gr = [](long long n) { return (int)((n + 255) / 256); };

  // ---- softmaxed causal matrices ----
  row_softmax<<<1, OBS_, 0, stream>>>(caus, smxC, OBS_, OBS_);
  row_softmax<<<1, N_, 0, stream>>>(scmC, smxN, N_, N_);

  // ---- plain GAT (concat=False -> head mean) ----
  gemm(obs, gatW, nullptr, xt, B_ * N_, GD_, OBS_,
       OBS_, GD_, GD_, 0, (long long)OBS_ * GD_, (long long)B_ * N_ * GD_, 0, H_, 0);
  gat_scores<<<gr((long long)H_ * B_ * N_), 256, 0, stream>>>(xt, gatA, s1b, s2b);
  gat_attn<<<B_ * H_, 256, 0, stream>>>(s1b, s2b, adj, xt, at,
                                        (long long)B_ * N_ * GD_, GD_, 0);
  head_mean<<<gr((long long)B_ * N_ * GD_), 256, 0, stream>>>(at, comm, 0);

  // ---- CausalGAT: xm = obs @ softmax(causal); 2 GAT layers, ELU between ----
  gemm(obs, smxC, nullptr, xm, B_ * N_, OBS_, OBS_, OBS_, OBS_, OBS_,
       0, 0, 0, 0, 1, 0);
  gemm(xm, cgW0, nullptr, xt, B_ * N_, GD_, OBS_,
       OBS_, GD_, GD_, 0, (long long)OBS_ * GD_, (long long)B_ * N_ * GD_, 0, H_, 0);
  gat_scores<<<gr((long long)H_ * B_ * N_), 256, 0, stream>>>(xt, cgA0, s1b, s2b);
  gat_attn<<<B_ * H_, 256, 0, stream>>>(s1b, s2b, adj, xt, hcg,
                                        GD_, H_ * GD_, 2 /*ELU, concat layout*/);
  gemm(hcg, cgW1, nullptr, xt, B_ * N_, GD_, H_ * GD_,
       H_ * GD_, GD_, GD_, 0, (long long)(H_ * GD_) * GD_,
       (long long)B_ * N_ * GD_, 0, H_, 0);
  gat_scores<<<gr((long long)H_ * B_ * N_), 256, 0, stream>>>(xt, cgA1, s1b, s2b);
  gat_attn<<<B_ * H_, 256, 0, stream>>>(s1b, s2b, adj, xt, at,
                                        (long long)B_ * N_ * GD_, GD_, 0);
  head_mean<<<gr((long long)B_ * N_ * GD_), 256, 0, stream>>>(at, comm, 1 /*add*/);

  // ---- ai = concat(obs, comm) ----  (hcg dead -> ai overlays it)
  concat2<<<B_ * N_, 256, 0, stream>>>(obs, comm, ai, OBS_, GD_);

  // ---- actor MLP (per-agent batched) ----
  gemm(ai, aw1, ab1, h1, B_, HID_, AI_,
       N_ * AI_, HID_, N_ * HID_,
       AI_, (long long)AI_ * HID_, HID_, HID_, N_, 1);
  gemm(h1, aw2, ab2, h2, B_, HID_, HID_,
       N_ * HID_, HID_, N_ * HID_,
       HID_, (long long)HID_ * HID_, HID_, HID_, N_, 1);
  gemm(h2, aw3, ab3, out_actor, B_, ACT_, HID_,
       N_ * HID_, ACT_, N_ * ACT_,
       HID_, (long long)HID_ * ACT_, ACT_, ACT_, N_, 0);

  // ---- critic MLP ----
  gemm(ai, kw1, kb1, h1, B_, HID_, AI_,
       N_ * AI_, HID_, N_ * HID_,
       AI_, (long long)AI_ * HID_, HID_, HID_, N_, 1);
  gemm(h1, kw2, kb2, h2, B_, HID_, HID_,
       N_ * HID_, HID_, N_ * HID_,
       HID_, (long long)HID_ * HID_, HID_, HID_, N_, 1);
  gemm(h2, kw3, kb3, out_critic, B_, 1, HID_,
       N_ * HID_, 1, N_,
       HID_, HID_, 1, 1, N_, 0);

  // ---- centralized critic over flat (obs||actions) ----
  concat2<<<B_ * N_, 256, 0, stream>>>(obs, acts, cmb, OBS_, ACT_);
  gemm(cmb, cw1, cb1, c1, B_, HID_, N_ * CI_,
       N_ * CI_, HID_, HID_, 0, 0, 0, 0, 1, 1);
  gemm(c1, cw2, cb2, c2, B_, HID_, HID_, HID_, HID_, HID_, 0, 0, 0, 0, 1, 1);
  gemm(c2, cw3, cb3, out_cent, B_, N_, HID_, HID_, N_, N_, 0, 0, 0, 0, 1, 0);

  // ---- SCM mechanisms: ce = mlp3(obs||actions) per-agent ----
  gemm(cmb, mw1, mb1, h1, B_, HID_, CI_,
       N_ * CI_, HID_, N_ * HID_,
       CI_, (long long)CI_ * HID_, HID_, HID_, N_, 1);
  gemm(h1, mw2, mb2, h2, B_, HID_, HID_,
       N_ * HID_, HID_, N_ * HID_,
       HID_, (long long)HID_ * HID_, HID_, HID_, N_, 1);
  gemm(h2, mw3, mb3, ce, B_, OBS_, HID_,
       N_ * HID_, OBS_, N_ * OBS_,
       HID_, (long long)HID_ * OBS_, OBS_, OBS_, N_, 0);
  // se[b] = softmax(scm_causal) @ ce[b]   (batched over B)
  gemm(smxN, ce, nullptr, se, N_, OBS_, N_,
       N_, OBS_, OBS_, 0, (long long)N_ * OBS_, (long long)N_ * OBS_, 0, B_, 0);
  // noise = mlp2(obs) per-agent
  gemm(obs, nw1, nb1, h1, B_, HID_, OBS_,
       N_ * OBS_, HID_, N_ * HID_,
       OBS_, (long long)OBS_ * HID_, HID_, HID_, N_, 1);
  gemm(h1, nw2, nb2, noise, B_, OBS_, HID_,
       N_ * HID_, OBS_, N_ * OBS_,
       HID_, (long long)HID_ * OBS_, OBS_, OBS_, N_, 0);
  add_vec<<<gr((long long)B_ * N_ * OBS_), 256, 0, stream>>>(se, noise, out_scm,
                                                             (long long)B_ * N_ * OBS_);
}
